// StockModel_14010183320166
// MI455X (gfx1250) — compile-verified, matrix operand order checked by reading, not run on hardware
//
#include <hip/hip_runtime.h>

// ---------------- model dims ----------------
constexpr int TT   = 4;
constexpr int VV   = 4096;
constexpr int EE   = 4096;
constexpr int KK   = 32;    // hyperedge members
constexpr int DD   = 8;     // edges per vertex
constexpr int HH   = 16;    // lstm hidden
constexpr int BERT = 768;
constexpr int CAT  = 784;   // HH + BERT
constexpr int KPAD = 800;   // CAT padded to multiple of 32 (25 wmma k-steps)
constexpr int N1   = 196;   // CAT/4
constexpr int N1P  = 224;   // padded to 14 n-tiles = 2 groups of 7
constexpr int ROWS = TT * EE;   // 16384 hyperedge rows == TT*VV vertex rows

// ---------------- wmma types ----------------
typedef __attribute__((ext_vector_type(16))) __bf16 v16bf;
typedef __attribute__((ext_vector_type(8)))  float  v8f;

union Frag { uint4 u[2]; v16bf v; };

// ---------------- helpers ----------------
__device__ __forceinline__ unsigned short f2bf(float f) {
    unsigned int u = __float_as_uint(f);
    unsigned int r = (u + 0x7FFFu + ((u >> 16) & 1u)) >> 16;
    return (unsigned short)r;
}
__device__ __forceinline__ float bf2f(unsigned short h) {
    return __uint_as_float(((unsigned int)h) << 16);
}
__device__ __forceinline__ float sigmoidf(float x) {
    return 1.0f / (1.0f + __expf(-x));
}

// ================= kernel 1: weight prep (fp32 -> padded bf16) =================
__global__ void prep_weights_kernel(const float* __restrict__ ec_w1,   // (196,784)
                                    const float* __restrict__ Wih_m,   // (64,784)
                                    unsigned short* __restrict__ w1b,  // (224,800)
                                    unsigned short* __restrict__ wmb)  // (64,800)
{
    int i = blockIdx.x * 256 + threadIdx.x;
    const int total1 = N1P * KPAD;            // 179200
    const int total2 = 64 * KPAD;             // 51200
    if (i < total1) {
        int r = i / KPAD, c = i % KPAD;
        float v = (r < N1 && c < CAT) ? ec_w1[r * CAT + c] : 0.0f;
        w1b[i] = f2bf(v);
    } else if (i < total1 + total2) {
        int j = i - total1;
        int r = j / KPAD, c = j % KPAD;
        float v = (c < CAT) ? Wih_m[r * CAT + c] : 0.0f;
        wmb[j] = f2bf(v);
    }
}

// ================= kernel 2: price LSTM (T,V,1) -> (T,V,16) =================
__global__ void lstm1_kernel(const float* __restrict__ prices,
                             const float* __restrict__ Wih,  // (64,1)
                             const float* __restrict__ Whh,  // (64,16)
                             const float* __restrict__ bih,
                             const float* __restrict__ bhh,
                             float* __restrict__ np_out)     // (T,V,16)
{
    __shared__ float sW[64];
    __shared__ float sU[64 * 16];
    __shared__ float sb[64];
    int tid = threadIdx.x;
    if (tid < 64) { sW[tid] = Wih[tid]; sb[tid] = bih[tid] + bhh[tid]; }
    for (int i = tid; i < 64 * 16; i += 256) sU[i] = Whh[i];
    __syncthreads();

    int v = blockIdx.x * 256 + tid;
    float h[HH], c[HH];
#pragma unroll
    for (int u = 0; u < HH; ++u) { h[u] = 0.0f; c[u] = 0.0f; }

    for (int t = 0; t < TT; ++t) {
        float x = prices[(size_t)t * VV + v];
        float g[64];
#pragma unroll 4
        for (int j = 0; j < 64; ++j) {
            float s = sW[j] * x + sb[j];
            const float* ur = sU + j * HH;
#pragma unroll
            for (int u = 0; u < HH; ++u) s += ur[u] * h[u];
            g[j] = s;
        }
#pragma unroll
        for (int u = 0; u < HH; ++u) {
            float ig = sigmoidf(g[u]);
            float fg = sigmoidf(g[16 + u]);
            float gg = tanhf(g[32 + u]);
            float og = sigmoidf(g[48 + u]);
            c[u] = fg * c[u] + ig * gg;
            h[u] = og * tanhf(c[u]);
            np_out[((size_t)t * VV + v) * HH + u] = h[u];
        }
    }
}

// ================= kernel 3: VertexConv -> he_cat (bf16, K-padded) =================
__global__ void vertexconv_kernel(const float* __restrict__ np_in,     // (T,V,16)
                                  const int*   __restrict__ members,   // (T,E,32)
                                  const float* __restrict__ w,         // (32,32,16)
                                  const float* __restrict__ b,         // (32,32)
                                  const float* __restrict__ cw,        // (32)
                                  const float* __restrict__ cb,        // (1)
                                  const float* __restrict__ node_embs, // (T,E,768)
                                  unsigned short* __restrict__ hecat)  // (T*E,800)
{
    __shared__ float reg[KK][HH];
    __shared__ float mlt[KK][KK];
    __shared__ float trs[KK][HH];
    int blk = blockIdx.x;           // t*E + e
    int t   = blk >> 12;
    int tid = threadIdx.x;

    const int* mem = members + (size_t)blk * KK;
    for (int i = tid; i < KK * HH; i += 256) {
        int k = i >> 4, d = i & 15;
        int m = mem[k];
        reg[k][d] = np_in[((size_t)t * VV + m) * HH + d];
    }
    __syncthreads();

    for (int i = tid; i < KK * KK; i += 256) {
        int k = i >> 5, j = i & 31;
        const float* wr = w + ((size_t)k * KK + j) * HH;
        float s = b[k * KK + j];
#pragma unroll
        for (int d = 0; d < HH; ++d) s += reg[k][d] * wr[d];
        mlt[k][j] = s;
    }
    __syncthreads();

    if (tid < KK) {                  // softmax over j per row k
        int k = tid;
        float mx = mlt[k][0];
        for (int j = 1; j < KK; ++j) mx = fmaxf(mx, mlt[k][j]);
        float sum = 0.0f;
        for (int j = 0; j < KK; ++j) { float e = __expf(mlt[k][j] - mx); mlt[k][j] = e; sum += e; }
        float inv = 1.0f / sum;
        for (int j = 0; j < KK; ++j) mlt[k][j] *= inv;
    }
    __syncthreads();

    for (int i = tid; i < KK * HH; i += 256) {
        int k = i >> 4, d = i & 15;
        float s = 0.0f;
#pragma unroll 8
        for (int j = 0; j < KK; ++j) s += mlt[k][j] * reg[j][d];
        trs[k][d] = s;
    }
    __syncthreads();

    unsigned short* row = hecat + (size_t)blk * KPAD;
    if (tid < HH) {
        float s = cb[0];
#pragma unroll 8
        for (int k = 0; k < KK; ++k) s += trs[k][tid] * cw[k];
        row[tid] = f2bf(s);
    }
    const float* ne = node_embs + (size_t)blk * BERT;
    for (int c = tid; c < BERT; c += 256) row[HH + c] = f2bf(ne[c]);
    if (tid >= HH && tid < 2 * HH) row[CAT + (tid - HH)] = 0;   // zero k-pad
}

// ========== bf16 WMMA GEMM: wave per (mtile, n-group of NT tiles), compile-time strides ==========
// C = A(M x LDA) * B^T; B rows (n-tiles*16 x LDB) are W rows. A fragment loaded once per k-step and
// reused across NT WMMAs. Constant LDB folds all B-tile addresses into 24-bit instruction offsets
// (single base pointer, no per-tile pointer chains -> no spills). amdgpu_waves_per_eu(1) lets the
// allocator keep NT*8 accumulator VGPRs resident. Fragment packing per CDNA5 ISA 7.12.2.
template <int NT, int LDA, int LDB>
__global__ __attribute__((amdgpu_waves_per_eu(1)))
void gemm_bf16_wmma_nt(const unsigned short* __restrict__ A,
                       const unsigned short* __restrict__ B,
                       float* __restrict__ C, int ldc,
                       int mTiles, int nGroups, int kSteps,
                       const float* __restrict__ bias, int nBias, int doRelu)
{
    int gwave = (blockIdx.x * blockDim.x + threadIdx.x) >> 5;
    if (gwave >= mTiles * nGroups) return;
    int mtile = gwave / nGroups;
    int ng    = gwave % nGroups;

    int lane = threadIdx.x & 31;
    int r    = lane & 15;
    int half = lane >> 4;

    const unsigned short* aptr = A + (size_t)(mtile * 16 + r) * LDA + half * 8;
    const unsigned short* bptr = B + ((size_t)ng * NT * 16 + r) * LDB + half * 8;

    v8f acc[NT];
#pragma unroll
    for (int n = 0; n < NT; ++n) acc[n] = (v8f){0.f, 0.f, 0.f, 0.f, 0.f, 0.f, 0.f, 0.f};

    for (int ks = 0; ks < kSteps; ++ks) {
        Frag fa;
        fa.u[0] = *(const uint4*)(aptr);        // K = k0..k0+7
        fa.u[1] = *(const uint4*)(aptr + 16);   // K = k0+16..k0+23
#pragma unroll
        for (int n = 0; n < NT; ++n) {
            Frag fb;
            fb.u[0] = *(const uint4*)(bptr + n * 16 * LDB);        // immediate offsets
            fb.u[1] = *(const uint4*)(bptr + n * 16 * LDB + 16);
            acc[n] = __builtin_amdgcn_wmma_f32_16x16x32_bf16(
                false, fa.v, false, fb.v, (short)0, acc[n], false, false);
        }
        aptr += 32;
        bptr += 32;
    }

    int mbase   = mtile * 16 + half * 8;
    int colbase = ng * NT * 16 + r;
#pragma unroll
    for (int n = 0; n < NT; ++n) {
        int col = colbase + n * 16;
        float badd = (bias != nullptr && col < nBias) ? bias[col] : 0.0f;
#pragma unroll
        for (int rr = 0; rr < 8; ++rr) {
            float val = acc[n][rr] + badd;
            if (doRelu) val = fmaxf(val, 0.0f);
            C[(size_t)(mbase + rr) * ldc + col] = val;
        }
    }
}

// ================= kernel 5: per-hyperedge score (hidden @ ec_w2^T + b2) =================
__global__ void edge_score_kernel(const float* __restrict__ hidden,  // (ROWS,224)
                                  const float* __restrict__ w2,      // (196)
                                  const float* __restrict__ b2,
                                  float* __restrict__ sedge)         // (ROWS)
{
    int i = blockIdx.x * blockDim.x + threadIdx.x;
    if (i >= ROWS) return;
    const float* hr = hidden + (size_t)i * N1P;
    float s = b2[0];
#pragma unroll 4
    for (int n = 0; n < N1; ++n) s += hr[n] * w2[n];
    sedge[i] = s;
}

// ================= kernel 6: edge softmax-aggregate -> hg (bf16) =================
__global__ void edge_agg_kernel(const int* __restrict__ vert_edges,     // (T,V,8)
                                const float* __restrict__ sedge,        // (T*E)
                                const unsigned short* __restrict__ hecat,
                                unsigned short* __restrict__ hg)        // (T*V,800)
{
    __shared__ int   idx[DD];
    __shared__ float w[DD];
    int blk = blockIdx.x;          // t*V + v
    int t   = blk >> 12;
    int tid = threadIdx.x;
    if (tid < DD) idx[tid] = vert_edges[(size_t)blk * DD + tid];
    __syncthreads();
    if (tid == 0) {
        float s[DD], mx = -1e30f;
        for (int d = 0; d < DD; ++d) { s[d] = sedge[(size_t)t * EE + idx[d]]; mx = fmaxf(mx, s[d]); }
        float sum = 0.0f;
        for (int d = 0; d < DD; ++d) { s[d] = __expf(s[d] - mx); sum += s[d]; }
        float inv = 1.0f / sum;
        for (int d = 0; d < DD; ++d) w[d] = s[d] * inv;
    }
    __syncthreads();
    size_t tE = (size_t)t * EE;
    unsigned short* orow = hg + (size_t)blk * KPAD;
    for (int c = tid; c < KPAD; c += 256) {
        float acc = 0.0f;
#pragma unroll
        for (int d = 0; d < DD; ++d)
            acc += w[d] * bf2f(hecat[(tE + idx[d]) * KPAD + c]);
        orow[c] = f2bf(acc);
    }
}

// ================= kernel 8: LSTM2 recurrence + residual + attention + FC =================
__global__ void lstm2_attn_kernel(const float* __restrict__ xg,       // (T*V,64) precomputed x@Wih^T
                                  const float* __restrict__ Whh,      // (64,16)
                                  const float* __restrict__ bih,
                                  const float* __restrict__ bhh,
                                  const float* __restrict__ np_in,    // (T,V,16)
                                  const float* __restrict__ attn_in,  // (16,16)
                                  const float* __restrict__ attn_out, // (16,32)
                                  const float* __restrict__ fc_w,     // (2,16)
                                  const float* __restrict__ fc_b,     // (2)
                                  float* __restrict__ out)            // (V,2)
{
    __shared__ float sU[64 * 16];
    __shared__ float sb[64];
    int tid = threadIdx.x;
    if (tid < 64) sb[tid] = bih[tid] + bhh[tid];
    for (int i = tid; i < 64 * 16; i += 256) sU[i] = Whh[i];
    __syncthreads();

    int v = blockIdx.x * 256 + tid;
    float h[HH], c[HH], la[TT][HH];
#pragma unroll
    for (int u = 0; u < HH; ++u) { h[u] = 0.0f; c[u] = 0.0f; }

    for (int t = 0; t < TT; ++t) {
        size_t row = (size_t)t * VV + v;
        const float* xr = xg + row * 64;
        float g[64];
#pragma unroll 4
        for (int j = 0; j < 64; ++j) {
            float s = xr[j] + sb[j];
            const float* ur = sU + j * HH;
#pragma unroll
            for (int u = 0; u < HH; ++u) s += ur[u] * h[u];
            g[j] = s;
        }
#pragma unroll
        for (int u = 0; u < HH; ++u) {
            float ig = sigmoidf(g[u]);
            float fg = sigmoidf(g[16 + u]);
            float gg = tanhf(g[32 + u]);
            float og = sigmoidf(g[48 + u]);
            c[u] = fg * c[u] + ig * gg;
            h[u] = og * tanhf(c[u]);
            la[t][u] = h[u] + np_in[row * HH + u];
        }
    }

    // general attention: q = la[3] @ attn_in^T
    float q[HH];
#pragma unroll
    for (int u = 0; u < HH; ++u) {
        float s = 0.0f;
        const float* ar = attn_in + u * HH;
#pragma unroll
        for (int d = 0; d < HH; ++d) s += ar[d] * la[TT - 1][d];
        q[u] = s;
    }
    float sc[TT], mx = -1e30f;
#pragma unroll
    for (int t = 0; t < TT; ++t) {
        float s = 0.0f;
#pragma unroll
        for (int d = 0; d < HH; ++d) s += q[d] * la[t][d];
        sc[t] = s; mx = fmaxf(mx, s);
    }
    float sum = 0.0f;
#pragma unroll
    for (int t = 0; t < TT; ++t) { sc[t] = __expf(sc[t] - mx); sum += sc[t]; }
    float inv = 1.0f / sum;
    float mixv[HH];
#pragma unroll
    for (int d = 0; d < HH; ++d) {
        float s = 0.0f;
#pragma unroll
        for (int t = 0; t < TT; ++t) s += sc[t] * inv * la[t][d];
        mixv[d] = s;
    }
    float a[HH];
#pragma unroll
    for (int u = 0; u < HH; ++u) {
        const float* wr = attn_out + u * 2 * HH;
        float s = 0.0f;
#pragma unroll
        for (int j = 0; j < HH; ++j) s += wr[j] * mixv[j];
#pragma unroll
        for (int j = 0; j < HH; ++j) s += wr[HH + j] * q[j];
        a[u] = tanhf(s);
    }
    float o0 = fc_b[0], o1 = fc_b[1];
#pragma unroll
    for (int u = 0; u < HH; ++u) { o0 += fc_w[u] * a[u]; o1 += fc_w[HH + u] * a[u]; }
    out[(size_t)v * 2 + 0] = o0;
    out[(size_t)v * 2 + 1] = o1;
}

// ================= host launcher =================
extern "C" void kernel_launch(void* const* d_in, const int* in_sizes, int n_in,
                              void* d_out, int out_size, void* d_ws, size_t ws_size,
                              hipStream_t stream) {
    (void)in_sizes; (void)n_in; (void)out_size; (void)ws_size;

    const float* prices     = (const float*)d_in[0];
    const float* node_embs  = (const float*)d_in[1];
    const int*   he_members = (const int*)d_in[2];
    const int*   vert_edges = (const int*)d_in[3];
    const float* Wih_p = (const float*)d_in[4];
    const float* Whh_p = (const float*)d_in[5];
    const float* bih_p = (const float*)d_in[6];
    const float* bhh_p = (const float*)d_in[7];
    const float* Wih_m = (const float*)d_in[8];
    const float* Whh_m = (const float*)d_in[9];
    const float* bih_m = (const float*)d_in[10];
    const float* bhh_m = (const float*)d_in[11];
    const float* vc_trans_w = (const float*)d_in[12];
    const float* vc_trans_b = (const float*)d_in[13];
    const float* vc_conv_w  = (const float*)d_in[14];
    const float* vc_conv_b  = (const float*)d_in[15];
    const float* ec_w1 = (const float*)d_in[16];
    const float* ec_b1 = (const float*)d_in[17];
    const float* ec_w2 = (const float*)d_in[18];
    const float* ec_b2 = (const float*)d_in[19];
    const float* attn_in  = (const float*)d_in[20];
    const float* attn_out = (const float*)d_in[21];
    const float* fc_w = (const float*)d_in[22];
    const float* fc_b = (const float*)d_in[23];

    char* ws = (char*)d_ws;
    size_t off = 0;
    auto bump = [&](size_t bytes) -> void* {
        void* p = ws + off;
        off = (off + bytes + 255) & ~(size_t)255;
        return p;
    };
    float*          np_buf = (float*)bump((size_t)TT * VV * HH * 4);           // 1 MB
    unsigned short* hecat  = (unsigned short*)bump((size_t)ROWS * KPAD * 2);   // 26 MB
    unsigned short* w1b    = (unsigned short*)bump((size_t)N1P * KPAD * 2);
    unsigned short* wmb    = (unsigned short*)bump((size_t)64 * KPAD * 2);
    float*          hidden = (float*)bump((size_t)ROWS * N1P * 4);             // 14.7 MB
    float*          sedge  = (float*)bump((size_t)ROWS * 4);
    unsigned short* hg     = (unsigned short*)bump((size_t)ROWS * KPAD * 2);   // 26 MB
    float*          xg     = (float*)bump((size_t)ROWS * 64 * 4);              // 4 MB

    // 1) weight conversion/padding
    {
        int total = N1P * KPAD + 64 * KPAD;
        prep_weights_kernel<<<(total + 255) / 256, 256, 0, stream>>>(ec_w1, Wih_m, w1b, wmb);
    }
    // 2) price LSTM
    lstm1_kernel<<<VV / 256, 256, 0, stream>>>(prices, Wih_p, Whh_p, bih_p, bhh_p, np_buf);
    // 3) VertexConv -> he_cat (bf16)
    vertexconv_kernel<<<TT * EE, 256, 0, stream>>>(np_buf, he_members, vc_trans_w, vc_trans_b,
                                                   vc_conv_w, vc_conv_b, node_embs, hecat);
    // 4) hidden = relu(he_cat @ ec_w1^T + b1)  [WMMA bf16, M=16384 N=224 K=800, NT=7 x 2 groups]
    {
        int mTiles = ROWS / 16, nGroups = 2;   // 2048 waves
        gemm_bf16_wmma_nt<7, KPAD, KPAD><<<mTiles * nGroups * 32 / 256, 256, 0, stream>>>(
            hecat, w1b, hidden, N1P, mTiles, nGroups, KPAD / 32, ec_b1, N1, 1);
    }
    // 5) per-edge scalar score
    edge_score_kernel<<<ROWS / 256, 256, 0, stream>>>(hidden, ec_w2, ec_b2, sedge);
    // 6) per-vertex softmax over D edges + weighted aggregate -> hg (bf16)
    edge_agg_kernel<<<TT * VV, 256, 0, stream>>>(vert_edges, sedge, hecat, hg);
    // 7) xg = hg @ Wih_m^T  [WMMA bf16, M=16384 N=64 K=800, NT=4 x 1 group]
    {
        int mTiles = ROWS / 16, nGroups = 1;   // 1024 waves
        gemm_bf16_wmma_nt<4, KPAD, KPAD><<<mTiles * nGroups * 32 / 256, 256, 0, stream>>>(
            hg, wmb, xg, 64, mTiles, nGroups, KPAD / 32, nullptr, 0, 0);
    }
    // 8) LSTM2 + residual + attention + classifier
    lstm2_attn_kernel<<<VV / 256, 256, 0, stream>>>(xg, Whh_m, bih_m, bhh_m, np_buf,
                                                    attn_in, attn_out, fc_w, fc_b, (float*)d_out);
}
